// GuidedAttention_64931315581178
// MI455X (gfx1250) — compile-verified, matrix operand order checked by reading, not run on hardware
//
#include <hip/hip_runtime.h>
#include <hip/hip_bf16.h>

// ---------------------------------------------------------------------------
// Guided multi-head attention for MI455X (gfx1250), wave32 + WMMA bf16.
//   B=8, Ta=1024, Tv=512, D=1024, H=16, hd=64
// Pass 0: one-shot f32->bf16 conversion of activations/weights (BW-bound).
// GEMMs:  pure-bf16 b128 operand loads, software-pipelined (next-iteration
//         fragments prefetched while current WMMAs issue), launch_bounds to
//         keep everything in registers (no scratch spills).
// Attn:   flash, 32 q-rows/wave; base-2 softmax via raw v_log_f32/v_exp_f32
//         builtins; score D-fragment feeds the P*V B-fragment (same-lane
//         repack, no shuffles).
// ---------------------------------------------------------------------------

typedef __attribute__((ext_vector_type(16))) __bf16 v16bf;
typedef __attribute__((ext_vector_type(8)))  __bf16 v8bf;
typedef __attribute__((ext_vector_type(8)))  float  v8f;

__device__ __forceinline__ v8f wmma_bf16(v16bf a, v16bf b, v8f c) {
  return __builtin_amdgcn_wmma_f32_16x16x32_bf16(false, a, false, b,
                                                 (short)0, c, false, false);
}

__device__ __forceinline__ float fast_log2(float x) {   // v_log_f32 (base-2)
  return __builtin_amdgcn_logf(x);
}
__device__ __forceinline__ float fast_exp2(float x) {   // v_exp_f32 (base-2)
  return __builtin_amdgcn_exp2f(x);
}

__device__ __forceinline__ v8f vzero8() {
  v8f z;
#pragma unroll
  for (int e = 0; e < 8; ++e) z[e] = 0.0f;
  return z;
}

// Fragment pointer for 16x32 bf16 tiles of a row-major [rows, ld] matrix.
// Lane l handles row (l&15); K elements are (kb..kb+7, kb+16..kb+23), kb=(l>=16)*8.
__device__ __forceinline__ const __bf16* frag_ptr(const __bf16* base, int ld) {
  const int l = threadIdx.x & 31;
  return base + (size_t)(l & 15) * ld + ((l >> 4) << 3);
}

__device__ __forceinline__ v16bf load_frag(const __bf16* p) {
  const v8bf lo = *(const v8bf*)p;
  const v8bf hi = *(const v8bf*)(p + 16);
  v16bf r;
#pragma unroll
  for (int e = 0; e < 8; ++e) {
    r[e]     = lo[e];
    r[8 + e] = hi[e];
  }
  return r;
}

// ---------------------------------------------------------------------------
// One-shot f32 -> bf16 conversion (8 elements/thread).
// ---------------------------------------------------------------------------
__global__ void cvt_bf16_kernel(const float* __restrict__ src, __bf16* __restrict__ dst,
                                int n) {
  const int i = (int)(blockIdx.x * blockDim.x + threadIdx.x) * 8;
  if (i >= n) return;
  const float4 a = *(const float4*)(src + i);
  const float4 b = *(const float4*)(src + i + 4);
  v8bf o;
  o[0] = (__bf16)a.x; o[1] = (__bf16)a.y; o[2] = (__bf16)a.z; o[3] = (__bf16)a.w;
  o[4] = (__bf16)b.x; o[5] = (__bf16)b.y; o[6] = (__bf16)b.z; o[7] = (__bf16)b.w;
  *(v8bf*)(dst + i) = o;
}

// ---------------------------------------------------------------------------
// Q/K projection: out[b,h,t,hd] = X[row,:] @ W[d,:]^T + bias[d]   (bf16 out)
// WMMA orientation: M = d (16-wide tile), N = rows (4x16 window per wave).
// Software-pipelined K loop: prefetch k+1 fragments, WMMA on k fragments.
// ---------------------------------------------------------------------------
template <int LOG2T>
__global__ __launch_bounds__(256, 1)
void proj_qk_kernel(const __bf16* __restrict__ X, const __bf16* __restrict__ W,
                    const float* __restrict__ bias, __bf16* __restrict__ out,
                    int rows) {
  const int T = 1 << LOG2T;
  const int lane = threadIdx.x & 31;
  const int wave = (int)((blockIdx.x * blockDim.x + threadIdx.x) >> 5);
  const int dt = wave & 63;   // 64 tiles of 16 output dims
  const int rw = wave >> 6;   // 64-row window
  if (rw * 64 >= rows) return;

  const __bf16* Wb = W + (size_t)dt * 16 * 1024;
  const __bf16* Xb = X + (size_t)(rw * 64) * 1024;

  v8f acc[4];
#pragma unroll
  for (int i = 0; i < 4; ++i) acc[i] = vzero8();

  v16bf a_c = load_frag(frag_ptr(Wb, 1024));
  v16bf b_c[4];
#pragma unroll
  for (int i = 0; i < 4; ++i)
    b_c[i] = load_frag(frag_ptr(Xb + (size_t)i * 16 * 1024, 1024));

  for (int kk = 0; kk < 1024; kk += 32) {
    const int kn = (kk + 32) & 1023;      // wrap: last prefetch discarded
    const v16bf a_n = load_frag(frag_ptr(Wb + kn, 1024));
    v16bf b_n[4];
#pragma unroll
    for (int i = 0; i < 4; ++i)
      b_n[i] = load_frag(frag_ptr(Xb + (size_t)i * 16 * 1024 + kn, 1024));

#pragma unroll
    for (int i = 0; i < 4; ++i) acc[i] = wmma_bf16(a_c, b_c[i], acc[i]);

    a_c = a_n;
#pragma unroll
    for (int i = 0; i < 4; ++i) b_c[i] = b_n[i];
  }

  const int off8 = (lane >> 4) << 3;          // 0 or 8
  const int d0   = dt * 16 + off8;
  const int hh   = d0 >> 6;
  const int hdi  = d0 & 63;
  float bloc[8];
#pragma unroll
  for (int e = 0; e < 8; ++e) bloc[e] = bias[d0 + e];

#pragma unroll
  for (int i = 0; i < 4; ++i) {
    const int row = rw * 64 + i * 16 + (lane & 15);
    const int bb = row >> LOG2T, tt = row & (T - 1);
    __bf16* dst = out + (((size_t)bb * 16 + hh) * T + tt) * 64 + hdi;
    v8bf o;
#pragma unroll
    for (int e = 0; e < 8; ++e) o[e] = (__bf16)(acc[i][e] + bloc[e]);
    *(v8bf*)dst = o;
  }
}

// ---------------------------------------------------------------------------
// V projection, stored transposed: outT[b,h,hd,tv].
// WMMA orientation: M = tv rows (16), N = d (4x16 window). Pipelined K loop.
// ---------------------------------------------------------------------------
template <int LOG2T>
__global__ __launch_bounds__(256, 1)
void proj_v_kernel(const __bf16* __restrict__ X, const __bf16* __restrict__ W,
                   const float* __restrict__ bias, __bf16* __restrict__ outT,
                   int rows) {
  const int T = 1 << LOG2T;
  const int lane = threadIdx.x & 31;
  const int wave = (int)((blockIdx.x * blockDim.x + threadIdx.x) >> 5);
  const int dw = wave & 15;   // 16 windows of 64 output dims
  const int mt = wave >> 4;   // 16-row tiles
  if (mt * 16 >= rows) return;

  const __bf16* Xb = X + (size_t)mt * 16 * 1024;
  const __bf16* Wb = W + (size_t)(dw * 64) * 1024;

  v8f acc[4];
#pragma unroll
  for (int i = 0; i < 4; ++i) acc[i] = vzero8();

  v16bf a_c = load_frag(frag_ptr(Xb, 1024));
  v16bf b_c[4];
#pragma unroll
  for (int i = 0; i < 4; ++i)
    b_c[i] = load_frag(frag_ptr(Wb + (size_t)i * 16 * 1024, 1024));

  for (int kk = 0; kk < 1024; kk += 32) {
    const int kn = (kk + 32) & 1023;
    const v16bf a_n = load_frag(frag_ptr(Xb + kn, 1024));
    v16bf b_n[4];
#pragma unroll
    for (int i = 0; i < 4; ++i)
      b_n[i] = load_frag(frag_ptr(Wb + (size_t)i * 16 * 1024 + kn, 1024));

#pragma unroll
    for (int i = 0; i < 4; ++i) acc[i] = wmma_bf16(a_c, b_c[i], acc[i]);

    a_c = a_n;
#pragma unroll
    for (int i = 0; i < 4; ++i) b_c[i] = b_n[i];
  }

  const int off8 = (lane >> 4) << 3;
  const int rowb = mt * 16 + off8;
  const int bb = rowb >> LOG2T, tvb = rowb & (T - 1);

#pragma unroll
  for (int i = 0; i < 4; ++i) {
    const int d = dw * 64 + i * 16 + (lane & 15);
    const float bv = bias[d];
    const int hh = d >> 6, hdi = d & 63;
    __bf16* dst = outT + (((size_t)bb * 16 + hh) * 64 + hdi) * T + tvb;
    v8bf o;
#pragma unroll
    for (int e = 0; e < 8; ++e) o[e] = (__bf16)(acc[i][e] + bv);
    *(v8bf*)dst = o;
  }
}

// ---------------------------------------------------------------------------
// Flash attention: one wave per (b, h, 32-row q block), online softmax over Tv.
// Base-2 domain: t = s*(0.125*log2e) + log2(g+1e-8)*scale; p = exp2(t - m2).
// (log2e*ln2 == 1, so multiplying log2(g) by `scale` directly is exact.)
// Scores oriented D[m=tv][n=ta]; the exp'd score fragment IS the B-fragment
// for the P*V WMMA (same-lane repack). Per-qi softmax keeps one p[] live.
// ---------------------------------------------------------------------------
__global__ __launch_bounds__(256, 1)
void attn_kernel(const __bf16* __restrict__ Qh, const __bf16* __restrict__ Kh,
                 const __bf16* __restrict__ Vt, const float* __restrict__ guide,
                 const float* __restrict__ scale_p, __bf16* __restrict__ ctx_out) {
  const int lane = threadIdx.x & 31;
  const int wave = (int)((blockIdx.x * blockDim.x + threadIdx.x) >> 5);
  const int qt2 = wave & 31;          // 32-row q blocks (32 per b,h)
  const int h   = (wave >> 5) & 15;
  const int b   = wave >> 9;
  const float sc = scale_p[0];
  const float kQ = 0.125f * 1.44269504089f;   // (1/sqrt(hd)) * log2(e)
  const int off8 = (lane >> 4) << 3;

  const int bh = b * 16 + h;
  const __bf16* Qbase = Qh + ((size_t)bh * 1024 + qt2 * 32) * 64;
  const __bf16* Kbase = Kh + (size_t)bh * 512 * 64;
  const __bf16* Vbase = Vt + (size_t)bh * 64 * 512;
  const float* Groq = guide + ((size_t)b * 1024 + qt2 * 32 + (lane & 15)) * 512;

  // Q as B-operand (n = ta), reused across all kv tiles. 2 q-tiles x 2 hd halves.
  v16bf qf[2][2];
#pragma unroll
  for (int qi = 0; qi < 2; ++qi) {
    qf[qi][0] = load_frag(frag_ptr(Qbase + (size_t)qi * 16 * 64 + 0, 64));
    qf[qi][1] = load_frag(frag_ptr(Qbase + (size_t)qi * 16 * 64 + 32, 64));
  }

  v8f acc[2][4];
#pragma unroll
  for (int qi = 0; qi < 2; ++qi)
#pragma unroll
    for (int i = 0; i < 4; ++i) acc[qi][i] = vzero8();
  float m_run[2] = {-3.0e38f, -3.0e38f};
  float l_run[2] = {0.0f, 0.0f};

  for (int tv0 = 0; tv0 < 512; tv0 += 32) {
    // K fragments (A-operand, m = tv), shared by both q tiles.
    v16bf kf[2][2];
#pragma unroll
    for (int cc = 0; cc < 2; ++cc) {
      const __bf16* Kt = Kbase + (size_t)(tv0 + cc * 16) * 64;
      kf[cc][0] = load_frag(frag_ptr(Kt + 0, 64));
      kf[cc][1] = load_frag(frag_ptr(Kt + 32, 64));
    }
    // V fragments issued now; consumed only after the softmax below.
    v16bf vf[4];
#pragma unroll
    for (int i = 0; i < 4; ++i)
      vf[i] = load_frag(frag_ptr(Vbase + (size_t)(i * 16) * 512 + tv0, 512));

    v16bf pb[2];
    float corr[2];
#pragma unroll
    for (int qi = 0; qi < 2; ++qi) {
      // Scores for both 16-wide tv chunks of this window.
      v8f sA = vzero8(), sB = vzero8();
      sA = wmma_bf16(kf[0][0], qf[qi][0], sA);
      sA = wmma_bf16(kf[0][1], qf[qi][1], sA);
      sB = wmma_bf16(kf[1][0], qf[qi][0], sB);
      sB = wmma_bf16(kf[1][1], qf[qi][1], sB);

      const float* gp = Groq + (size_t)qi * 16 * 512 + tv0 + off8;
      float p[2][8];
      float wmax = -3.0e38f;
#pragma unroll
      for (int e = 0; e < 8; ++e) {
        const float t0 = sA[e] * kQ + fast_log2(gp[e]      + 1e-8f) * sc;
        const float t1 = sB[e] * kQ + fast_log2(gp[16 + e] + 1e-8f) * sc;
        p[0][e] = t0;
        p[1][e] = t1;
        wmax = fmaxf(wmax, fmaxf(t0, t1));
      }
      wmax = fmaxf(wmax, __shfl_xor(wmax, 16, 32));
      const float m_new = fmaxf(m_run[qi], wmax);
      corr[qi] = fast_exp2(m_run[qi] - m_new);
      float rsum = 0.0f;
#pragma unroll
      for (int cc = 0; cc < 2; ++cc)
#pragma unroll
        for (int e = 0; e < 8; ++e) {
          p[cc][e] = fast_exp2(p[cc][e] - m_new);
          rsum += p[cc][e];
        }
      rsum += __shfl_xor(rsum, 16, 32);
      l_run[qi] = l_run[qi] * corr[qi] + rsum;
      m_run[qi] = m_new;

      // B-fragment of P for K=32 (tv0..tv0+31): same-lane repack, no shuffles.
#pragma unroll
      for (int e = 0; e < 8; ++e) {
        pb[qi][e]     = (__bf16)p[0][e];
        pb[qi][8 + e] = (__bf16)p[1][e];
      }
    }

#pragma unroll
    for (int i = 0; i < 4; ++i) {
#pragma unroll
      for (int qi = 0; qi < 2; ++qi) {
#pragma unroll
        for (int e = 0; e < 8; ++e) acc[qi][i][e] *= corr[qi];
        acc[qi][i] = wmma_bf16(vf[i], pb[qi], acc[qi][i]);
      }
    }
  }

#pragma unroll
  for (int qi = 0; qi < 2; ++qi) {
    const float inv_l = 1.0f / l_run[qi];
    const int ta = qt2 * 32 + qi * 16 + (lane & 15);
    __bf16* obase = ctx_out + ((size_t)b * 1024 + ta) * 1024 + h * 64;
#pragma unroll
    for (int i = 0; i < 4; ++i) {
      v8bf o;
#pragma unroll
      for (int e = 0; e < 8; ++e) o[e] = (__bf16)(acc[qi][i][e] * inv_l);
      *(v8bf*)(obase + i * 16 + off8) = o;
    }
  }
}

// ---------------------------------------------------------------------------
// Out-projection + gated residual: x = q + gate*(ctx @ Wout^T + bout)  (f32)
// Pipelined K loop.
// ---------------------------------------------------------------------------
__global__ __launch_bounds__(256, 1)
void outproj_kernel(const __bf16* __restrict__ C, const __bf16* __restrict__ W,
                    const float* __restrict__ bias, const float* __restrict__ q,
                    const float* __restrict__ gate, float* __restrict__ Xo) {
  const int lane = threadIdx.x & 31;
  const int wave = (int)((blockIdx.x * blockDim.x + threadIdx.x) >> 5);
  const int dt = wave & 63;
  const int rw = wave >> 6;

  const __bf16* Wb = W + (size_t)dt * 16 * 1024;
  const __bf16* Cb = C + (size_t)(rw * 64) * 1024;

  v8f acc[4];
#pragma unroll
  for (int i = 0; i < 4; ++i) acc[i] = vzero8();

  v16bf a_c = load_frag(frag_ptr(Wb, 1024));
  v16bf b_c[4];
#pragma unroll
  for (int i = 0; i < 4; ++i)
    b_c[i] = load_frag(frag_ptr(Cb + (size_t)i * 16 * 1024, 1024));

  for (int kk = 0; kk < 1024; kk += 32) {
    const int kn = (kk + 32) & 1023;
    const v16bf a_n = load_frag(frag_ptr(Wb + kn, 1024));
    v16bf b_n[4];
#pragma unroll
    for (int i = 0; i < 4; ++i)
      b_n[i] = load_frag(frag_ptr(Cb + (size_t)i * 16 * 1024 + kn, 1024));

#pragma unroll
    for (int i = 0; i < 4; ++i) acc[i] = wmma_bf16(a_c, b_c[i], acc[i]);

    a_c = a_n;
#pragma unroll
    for (int i = 0; i < 4; ++i) b_c[i] = b_n[i];
  }

  const float g = gate[0];
  const int off8 = (lane >> 4) << 3;
  const int d0   = dt * 16 + off8;
  float bloc[8];
#pragma unroll
  for (int e = 0; e < 8; ++e) bloc[e] = bias[d0 + e];

#pragma unroll
  for (int i = 0; i < 4; ++i) {
    const int row = rw * 64 + i * 16 + (lane & 15);
    const float* qp = q + (size_t)row * 1024 + d0;
    float* dst = Xo + (size_t)row * 1024 + d0;
#pragma unroll
    for (int e = 0; e < 8; ++e) dst[e] = qp[e] + g * (acc[i][e] + bloc[e]);
  }
}

// ---------------------------------------------------------------------------
// LayerNorm over D=1024, one wave per row (32 elems/lane).
// ---------------------------------------------------------------------------
__global__ void ln_kernel(const float* __restrict__ X, const float* __restrict__ gamma,
                          const float* __restrict__ beta, float* __restrict__ out) {
  const int lane = threadIdx.x & 31;
  const int row  = (int)((blockIdx.x * blockDim.x + threadIdx.x) >> 5);
  const float* xr = X + (size_t)row * 1024;

  float vals[32];
  float s = 0.0f, s2 = 0.0f;
#pragma unroll
  for (int i = 0; i < 32; ++i) {
    const float x = xr[lane + i * 32];
    vals[i] = x;
    s += x;
    s2 += x * x;
  }
#pragma unroll
  for (int o = 16; o >= 1; o >>= 1) {
    s  += __shfl_xor(s, o, 32);
    s2 += __shfl_xor(s2, o, 32);
  }
  const float mu  = s * (1.0f / 1024.0f);
  const float var = s2 * (1.0f / 1024.0f) - mu * mu;
  const float rs  = rsqrtf(var + 1e-5f);

  float* orow = out + (size_t)row * 1024;
#pragma unroll
  for (int i = 0; i < 32; ++i) {
    const int c = lane + i * 32;
    orow[c] = (vals[i] - mu) * rs * gamma[c] + beta[c];
  }
}

// ---------------------------------------------------------------------------
extern "C" void kernel_launch(void* const* d_in, const int* in_sizes, int n_in,
                              void* d_out, int out_size, void* d_ws, size_t ws_size,
                              hipStream_t stream) {
  const float* q      = (const float*)d_in[0];
  const float* k      = (const float*)d_in[1];
  const float* v      = (const float*)d_in[2];
  const float* guide  = (const float*)d_in[3];
  const float* in_w   = (const float*)d_in[4];   // [3*1024, 1024] rows [Wq; Wk; Wv]
  const float* in_b   = (const float*)d_in[5];
  const float* out_w  = (const float*)d_in[6];
  const float* out_b  = (const float*)d_in[7];
  const float* gamma  = (const float*)d_in[8];
  const float* beta   = (const float*)d_in[9];
  const float* gate   = (const float*)d_in[10];
  const float* scale  = (const float*)d_in[11];

  char* ws = (char*)d_ws;
  __bf16* Qh   = (__bf16*)(ws + 0);          // [8,16,1024,64]  16.8 MB
  __bf16* Kh   = (__bf16*)(ws + 16777216);   // [8,16,512,64]    8.4 MB
  __bf16* Vt   = (__bf16*)(ws + 25165824);   // [8,16,64,512]    8.4 MB
  __bf16* Ctx  = (__bf16*)(ws + 33554432);   // [8192,1024]     16.8 MB
  float*  Xr   = (float*)(ws + 50331648);    // [8192,1024] f32 33.6 MB (late)
  // bf16 staging: dead after projections, aliases the Xr region (written later)
  __bf16* qbf  = (__bf16*)(ws + 50331648);   // 16.8 MB
  __bf16* kbf  = (__bf16*)(ws + 67108864);   //  8.4 MB
  __bf16* vbf  = (__bf16*)(ws + 75497472);   //  8.4 MB
  __bf16* wbf  = (__bf16*)(ws + 83886080);   // in_proj bf16 6.3 MB
  __bf16* owbf = (__bf16*)(ws + 90177536);   // out_w   bf16 2.1 MB

  // Pass 0: one-shot down-conversion (bandwidth-bound).
  cvt_bf16_kernel<<<4096, 256, 0, stream>>>(q,     qbf,  8 * 1024 * 1024);
  cvt_bf16_kernel<<<2048, 256, 0, stream>>>(k,     kbf,  4 * 1024 * 1024);
  cvt_bf16_kernel<<<2048, 256, 0, stream>>>(v,     vbf,  4 * 1024 * 1024);
  cvt_bf16_kernel<<<1536, 256, 0, stream>>>(in_w,  wbf,  3 * 1024 * 1024);
  cvt_bf16_kernel<<< 512, 256, 0, stream>>>(out_w, owbf, 1024 * 1024);

  // Projections (pure bf16 WMMA loops, software-pipelined)
  proj_qk_kernel<10><<<1024, 256, 0, stream>>>(qbf, wbf,                   in_b,        Qh, 8192);
  proj_qk_kernel<9> <<< 512, 256, 0, stream>>>(kbf, wbf + 1024 * 1024,     in_b + 1024, Kh, 4096);
  proj_v_kernel<9>  <<< 512, 256, 0, stream>>>(vbf, wbf + 2 * 1024 * 1024, in_b + 2048, Vt, 4096);

  // Flash attention with log-guide bias (8*16*32 = 4096 waves)
  attn_kernel<<<512, 256, 0, stream>>>(Qh, Kh, Vt, guide, scale, Ctx);

  // Output projection + gated residual, then LayerNorm
  outproj_kernel<<<1024, 256, 0, stream>>>(Ctx, owbf, out_b, q, gate, Xr);
  ln_kernel<<<1024, 256, 0, stream>>>(Xr, gamma, beta, (float*)d_out);
}
